// BidRNN_52080773431934
// MI455X (gfx1250) — compile-verified, matrix operand order checked by reading, not run on hardware
//
#include <hip/hip_runtime.h>
#include <hip/hip_bf16.h>
#include <math.h>

// B=64, T=512, I=512, H=1024 bidirectional RNN.
// prep  : f32->bf16 of x and all weights (one pass), zero barrier words.
// phase1: xp = x @ W_xh + b (bf16 WMMA, f32 accum) -> written into d_out slots.
//         WG tile 64x32, each wave: one 16-row M-tile x two 16-col N-subtiles
//         (A fragment reused for 2 WMMAs).
// phase2: persistent recurrence, 512 steps of h = tanh(h @ W_hh + xp).
//         W_hh slice (1024x16 bf16 = 32KB) LDS-resident across all steps,
//         h double-buffered in ws, one software grid barrier per step per dir.

#define BB 64
#define TT 512
#define II 512
#define HH 1024

typedef __attribute__((ext_vector_type(16))) __bf16   v16bf;
typedef __attribute__((ext_vector_type(8)))  float    v8f;
typedef __attribute__((ext_vector_type(4)))  unsigned u32x4;

union Frag16 {
    v16bf          v;
    __bf16         e[16];
    unsigned short s[16];
    u32x4          q[2];
};

__device__ __forceinline__ unsigned short f32_to_bf16_bits(float f) {
    union { __bf16 h; unsigned short s; } u;
    u.h = (__bf16)f;
    return u.s;
}

// ---------------------------------------------------------------------------
// Prep: x and the four weight matrices f32 -> bf16; zero barrier control.
// Pure streaming (~100 MB) -> a few microseconds at HBM rate.
// ---------------------------------------------------------------------------
__global__ void prep_kernel(const float* __restrict__ x32,
                            const float* __restrict__ wxf32, const float* __restrict__ wxb32,
                            const float* __restrict__ whf32, const float* __restrict__ whb32,
                            unsigned short* __restrict__ xbf,
                            unsigned short* __restrict__ wxf, unsigned short* __restrict__ wxb,
                            unsigned short* __restrict__ whf, unsigned short* __restrict__ whb,
                            unsigned* __restrict__ ctrl)
{
    const int stride = gridDim.x * blockDim.x;
    const int i = blockIdx.x * blockDim.x + threadIdx.x;
    if (i < 64) ctrl[i] = 0u;
    for (int k = i; k < BB * TT * II; k += stride) xbf[k] = f32_to_bf16_bits(x32[k]);
    for (int k = i; k < II * HH; k += stride) {
        wxf[k] = f32_to_bf16_bits(wxf32[k]);
        wxb[k] = f32_to_bf16_bits(wxb32[k]);
    }
    for (int k = i; k < HH * HH; k += stride) {
        whf[k] = f32_to_bf16_bits(whf32[k]);
        whb[k] = f32_to_bf16_bits(whb32[k]);
    }
}

// ---------------------------------------------------------------------------
// Phase 1: out[(b*T+t)*2H + dir*H + n] = x[b,t,:] @ W_xh[dir] + b_h[dir].
// Grid (512, 32, 2); WG = 128 thr (4 waves); tile M=64, N=32.
// ---------------------------------------------------------------------------
__global__ void __launch_bounds__(128)
xproj_kernel(const unsigned short* __restrict__ xbf,
             const unsigned short* __restrict__ wx_f,
             const unsigned short* __restrict__ wx_b,
             const float* __restrict__ bh_f, const float* __restrict__ bh_b,
             float* __restrict__ out)
{
    __shared__ u32x4 ldsB[16 * 2 * 64];   // [kb][nsub][qi*32+lane] = 32 KB

    const int dir = blockIdx.z;
    const int n0  = blockIdx.y * 32;
    const int m0  = blockIdx.x * 64;
    const int tid = threadIdx.x;
    const int w   = tid >> 5;
    const int l   = tid & 31;
    const int col  = l & 15;
    const int base = (l >> 4) * 8;     // K split across lane halves (ISA 7.12.2)

    const unsigned short* W  = dir ? wx_b : wx_f;
    const float*          bh = dir ? bh_b : bh_f;

    // Swizzle B (K x 32 slice, row-major) into fragment layout.
    // 32 fragment-sets (16 kb x 2 nsub); thread covers idx = 4*i + w.
    for (int i = 0; i < 8; ++i) {
        const int idx = 4 * i + w;
        const int kb  = idx >> 1;
        const int ns  = idx & 1;
        const int nc  = n0 + ns * 16 + col;
        Frag16 f;
#pragma unroll
        for (int j = 0; j < 8; ++j) {
            f.s[j]     = W[(kb * 32 + base + j)      * HH + nc];
            f.s[8 + j] = W[(kb * 32 + 16 + base + j) * HH + nc];
        }
        ldsB[(kb * 2 + ns) * 64 +      l] = f.q[0];
        ldsB[(kb * 2 + ns) * 64 + 32 + l] = f.q[1];
    }
    __syncthreads();

    const float bv0 = bh[n0 + col];
    const float bv1 = bh[n0 + 16 + col];
    v8f acc0 = { bv0, bv0, bv0, bv0, bv0, bv0, bv0, bv0 };
    v8f acc1 = { bv1, bv1, bv1, bv1, bv1, bv1, bv1, bv1 };

    const int             arow = m0 + 16 * w + (l & 15);
    const unsigned short* xr   = xbf + (size_t)arow * II;

    for (int kb = 0; kb < 16; ++kb) {
        Frag16 a;                                 // 2x global_load_b128
        a.q[0] = *(const u32x4*)(xr + kb * 32 + base);
        a.q[1] = *(const u32x4*)(xr + kb * 32 + 16 + base);
        Frag16 b0, b1;
        b0.q[0] = ldsB[(kb * 2) * 64 + l];
        b0.q[1] = ldsB[(kb * 2) * 64 + 32 + l];
        b1.q[0] = ldsB[(kb * 2 + 1) * 64 + l];
        b1.q[1] = ldsB[(kb * 2 + 1) * 64 + 32 + l];
        acc0 = __builtin_amdgcn_wmma_f32_16x16x32_bf16(false, a.v, false, b0.v,
                                                       (short)0, acc0, false, false);
        acc1 = __builtin_amdgcn_wmma_f32_16x16x32_bf16(false, a.v, false, b1.v,
                                                       (short)0, acc1, false, false);
    }

#pragma unroll
    for (int r = 0; r < 8; ++r) {
        const int    gr  = m0 + 16 * w + r + 8 * (l >> 4);   // C/D layout row
        const size_t rowo = (size_t)gr * (2 * HH) + (size_t)dir * HH;
        out[rowo + n0 + col]      = acc0[r];
        out[rowo + n0 + 16 + col] = acc1[r];
    }
}

// ---------------------------------------------------------------------------
// Software grid barrier (per direction group of 64 WGs).
// ---------------------------------------------------------------------------
__device__ __forceinline__ void grid_barrier(unsigned* cnt, unsigned* gen,
                                             unsigned nwg, int tid)
{
    __threadfence();
    __syncthreads();
    if (tid == 0) {
        const unsigned g = __hip_atomic_load(gen, __ATOMIC_RELAXED, __HIP_MEMORY_SCOPE_AGENT);
        const unsigned old = __hip_atomic_fetch_add(cnt, 1u, __ATOMIC_ACQ_REL, __HIP_MEMORY_SCOPE_AGENT);
        if (old == nwg - 1u) {
            __hip_atomic_store(cnt, 0u, __ATOMIC_RELAXED, __HIP_MEMORY_SCOPE_AGENT);
            __hip_atomic_fetch_add(gen, 1u, __ATOMIC_RELEASE, __HIP_MEMORY_SCOPE_AGENT);
        } else {
            while (__hip_atomic_load(gen, __ATOMIC_ACQUIRE, __HIP_MEMORY_SCOPE_AGENT) == g) {
                __builtin_amdgcn_s_sleep(8);
            }
        }
    }
    __syncthreads();
    __threadfence();
}

// ---------------------------------------------------------------------------
// Phase 2: persistent recurrence. 128 WGs (64 fwd + 64 bwd), 16 columns each.
// ---------------------------------------------------------------------------
__global__ void __launch_bounds__(128)
rnn_scan_kernel(const float* __restrict__ h0_f, const float* __restrict__ h0_b,
                const unsigned short* __restrict__ wh_f,
                const unsigned short* __restrict__ wh_b,
                unsigned short* __restrict__ hbuf,   // 4 * (BB*HH) bf16
                float* __restrict__ out,             // xp in, h out (same slots)
                unsigned* __restrict__ ctrl)
{
    __shared__ u32x4 ldsB[32 * 64];   // 32 k-blocks * 32 lanes * 32B = 32 KB

    const int wg  = blockIdx.x;
    const int dir = wg >> 6;
    const int n0  = (wg & 63) * 16;
    const int tid = threadIdx.x;
    const int w   = tid >> 5;
    const int l   = tid & 31;
    const int col  = l & 15;
    const int base = (l >> 4) * 8;
    const int m0w  = 16 * w;          // this wave's batch-row tile

    const unsigned short* W  = dir ? wh_b : wh_f;
    const float*          h0 = dir ? h0_b : h0_f;
    unsigned* cnt = ctrl + dir * 2;
    unsigned* gen = ctrl + dir * 2 + 1;

    // One-time: W_hh slice (1024 x 16) -> LDS fragment layout, kb = w + 4*i.
    for (int i = 0; i < 8; ++i) {
        const int kb = w + 4 * i;
        Frag16 f;
#pragma unroll
        for (int j = 0; j < 8; ++j) {
            f.s[j]     = W[(kb * 32 + base + j)      * HH + n0 + col];
            f.s[8 + j] = W[(kb * 32 + 16 + base + j) * HH + n0 + col];
        }
        ldsB[kb * 64 +      l] = f.q[0];
        ldsB[kb * 64 + 32 + l] = f.q[1];
    }

    // One-time: h_prev (f32) -> bf16 into parity-0 buffer, our 16 columns.
    unsigned short* hb0 = hbuf + (size_t)(dir * 2) * (BB * HH);
    for (int i = tid; i < BB * 16; i += 128) {
        const int row = i >> 4, c = i & 15;
        hb0[row * HH + n0 + c] = f32_to_bf16_bits(h0[row * HH + n0 + c]);
    }
    __syncthreads();
    grid_barrier(cnt, gen, 64u, tid);

    for (int s = 0; s < TT; ++s) {
        const int t = dir ? (TT - 1 - s) : s;
        const unsigned short* hcur = hbuf + (size_t)(dir * 2 + (s & 1))       * (BB * HH);
        unsigned short*       hnxt = hbuf + (size_t)(dir * 2 + ((s + 1) & 1)) * (BB * HH);

        v8f acc = {};
        const unsigned short* hr = hcur + (size_t)(m0w + (l & 15)) * HH;
#pragma unroll 8
        for (int kb = 0; kb < 32; ++kb) {
            Frag16 a;   // two contiguous 16B chunks per lane: global_load_b128 x2
            a.q[0] = *(const u32x4*)(hr + kb * 32 + base);
            a.q[1] = *(const u32x4*)(hr + kb * 32 + 16 + base);
            Frag16 b;
            b.q[0] = ldsB[kb * 64 + l];
            b.q[1] = ldsB[kb * 64 + 32 + l];
            acc = __builtin_amdgcn_wmma_f32_16x16x32_bf16(false, a.v, false, b.v,
                                                          (short)0, acc, false, false);
        }

#pragma unroll
        for (int r = 0; r < 8; ++r) {
            const int    bb   = m0w + r + 8 * (l >> 4);
            const size_t oidx = ((size_t)bb * TT + t) * (2 * HH) + (size_t)dir * HH + n0 + col;
            const float  v    = tanhf(acc[r] + out[oidx]);   // xp read, h written back
            out[oidx] = v;
            hnxt[bb * HH + n0 + col] = f32_to_bf16_bits(v);
        }
        grid_barrier(cnt, gen, 64u, tid);
    }
}

// ---------------------------------------------------------------------------
extern "C" void kernel_launch(void* const* d_in, const int* in_sizes, int n_in,
                              void* d_out, int out_size, void* d_ws, size_t ws_size,
                              hipStream_t stream)
{
    (void)in_sizes; (void)n_in; (void)out_size; (void)ws_size;

    const float* x   = (const float*)d_in[0];  // (B,T,I)
    const float* h0f = (const float*)d_in[1];
    const float* h0b = (const float*)d_in[2];
    const float* Wxf = (const float*)d_in[3];  // (I,H)
    const float* Whf = (const float*)d_in[4];  // (H,H)
    const float* bhf = (const float*)d_in[5];
    const float* Wxb = (const float*)d_in[6];
    const float* Whb = (const float*)d_in[7];
    const float* bhb = (const float*)d_in[8];
    float* out = (float*)d_out;                // (B,T,2H)

    // Workspace layout (256B-aligned regions): ~39 MB total.
    char* ws = (char*)d_ws;
    unsigned*       ctrl = (unsigned*)ws;                         // 256 B
    unsigned short* wxf  = (unsigned short*)(ws + 256);           // 1 MB
    unsigned short* wxb  = wxf + (size_t)II * HH;                 // 1 MB
    unsigned short* whf  = wxb + (size_t)II * HH;                 // 2 MB
    unsigned short* whb  = whf + (size_t)HH * HH;                 // 2 MB
    unsigned short* hbuf = whb + (size_t)HH * HH;                 // 512 KB
    unsigned short* xbf  = hbuf + (size_t)4 * BB * HH;            // 32 MB

    prep_kernel<<<2048, 256, 0, stream>>>(x, Wxf, Wxb, Whf, Whb,
                                          xbf, wxf, wxb, whf, whb, ctrl);

    xproj_kernel<<<dim3((BB * TT) / 64, HH / 32, 2), 128, 0, stream>>>(
        xbf, wxf, wxb, bhf, bhb, out);

    rnn_scan_kernel<<<128, 128, 0, stream>>>(h0f, h0b, whf, whb, hbuf, out, ctrl);
}
